// GraphConvolution_15573551415441
// MI455X (gfx1250) — matrix operand order repaired and checked by reading, not compiled, not run on hardware
//
#include <hip/hip_runtime.h>

typedef __attribute__((ext_vector_type(2))) float v2f;
typedef __attribute__((ext_vector_type(4))) float v4f;
typedef __attribute__((ext_vector_type(8))) float v8f;
typedef __attribute__((ext_vector_type(8)))  __bf16 v8bf;
typedef __attribute__((ext_vector_type(16))) __bf16 v16bf;

#define NB        8
#define NNODES    2048
#define FDIM      128
#define KB        128                      // K chunk staged in LDS (kernel 2)
#define WAVES_PB  4
#define ROWS_PW   16
#define ROWS_PB   (WAVES_PB * ROWS_PW)     // 64 rows per block
#define BLK_THREADS (WAVES_PB * 32)        // 128 threads = 4 wave32

// ---------------------------------------------------------------------------
// fp32 -> (bf16 hi, bf16 lo) split: x ~= hi + lo, dropped term ~2^-18 * |x|
// ---------------------------------------------------------------------------
__device__ __forceinline__ void split_bf16(float f, __bf16& h, __bf16& l) {
  h = (__bf16)f;
  l = (__bf16)(f - (float)h);
}

union V16U { v16bf v; __bf16 e[16]; };

// ===========================================================================
// Kernel 1: support[b] = x[b] @ W   -- exact fp32 WMMA (small: ~0.54 GFLOP)
// ===========================================================================
// LDS B layout: [kq][n][4] floats; element (kk,n) at sW[(kk>>2)*512 + n*4 + (kk&3)]
__global__ __launch_bounds__(BLK_THREADS) void gcn_support_kernel(
    const float* __restrict__ x, const float* __restrict__ W,
    float* __restrict__ support) {
  __shared__ float sW[(FDIM / 4) * FDIM * 4];   // 64 KB: full 128x128 W
  const int tid  = threadIdx.x;
  const int b    = blockIdx.x >> 5;             // 32 row-blocks per batch
  const int rblk = blockIdx.x & 31;

#pragma unroll 4
  for (int idx = tid; idx < (FDIM / 4) * FDIM; idx += BLK_THREADS) {
    int kq = idx >> 7, n = idx & 127;
    v4f v;
    v.x = W[(4 * kq + 0) * FDIM + n];
    v.y = W[(4 * kq + 1) * FDIM + n];
    v.z = W[(4 * kq + 2) * FDIM + n];
    v.w = W[(4 * kq + 3) * FDIM + n];
    *(v4f*)&sW[kq * 512 + n * 4] = v;
  }
  __syncthreads();

  const int wave = tid >> 5, lane = tid & 31;
  const int n15 = lane & 15, half = lane >> 4;
  const int r0 = rblk * ROWS_PB + wave * ROWS_PW;

  v8f acc[8];
  const v8f vzero = {0.f, 0.f, 0.f, 0.f, 0.f, 0.f, 0.f, 0.f};
#pragma unroll
  for (int i = 0; i < 8; ++i) acc[i] = vzero;

  const float* aRow = x + ((size_t)b * NNODES + r0 + n15) * FDIM;
  for (int ks = 0; ks < FDIM / 4; ++ks) {
    v2f a = *(const v2f*)(aRow + ks * 4 + 2 * half);     // A: K{0,1}|{2,3}
    const float* bBase = sW + ks * 512 + 2 * half;
#pragma unroll
    for (int nt = 0; nt < 8; ++nt) {
      v2f bb = *(const v2f*)(bBase + (nt * 16 + n15) * 4);
      acc[nt] = __builtin_amdgcn_wmma_f32_16x16x4_f32(
          false, a, false, bb, (short)0, acc[nt], false, false);
    }
  }

  float* o = support + ((size_t)b * NNODES + r0) * FDIM;
#pragma unroll
  for (int nt = 0; nt < 8; ++nt) {
    const int col = nt * 16 + n15;
#pragma unroll
    for (int i = 0; i < 8; ++i)           // C layout: V[i] -> row i + 8*half
      o[(i + 8 * half) * FDIM + col] = acc[nt][i];
  }
}

// ===========================================================================
// Kernel 2: out[b] = adj[b] @ support[b] + bias  -- bf16x3 split WMMA
//   (hi*hi + lo*hi + hi*lo with fp32 accumulate: ~fp32 accuracy, 2.7x fewer
//    matrix ops than f32 WMMA, using the 16x16x32 bf16 XDL path)
// ===========================================================================
// LDS: two bf16 planes (hi, lo), K-contiguous per column, XOR-swizzled:
//   element (n, kk) at bf16 index  n*KB + ((kk>>3) ^ (n&15))*8 + (kk&7)
// A B-frag (K{0..7,16..23} (half 0) or {8..15,24..31} (half 1) of column n) = two aligned ds_load_b128.
__global__ __launch_bounds__(BLK_THREADS) void gcn_aggregate_kernel(
    const float* __restrict__ adj, const float* __restrict__ support,
    const float* __restrict__ bias, float* __restrict__ out) {
  __shared__ __bf16 sHI[KB * FDIM];   // 32 KB
  __shared__ __bf16 sLO[KB * FDIM];   // 32 KB
  const int tid  = threadIdx.x;
  const int b    = blockIdx.x >> 5;
  const int rblk = blockIdx.x & 31;
  const int wave = tid >> 5, lane = tid & 31;
  const int n15 = lane & 15, half = lane >> 4;
  const int r0 = rblk * ROWS_PB + wave * ROWS_PW;

  v8f acc[8];
  const v8f vzero = {0.f, 0.f, 0.f, 0.f, 0.f, 0.f, 0.f, 0.f};
#pragma unroll
  for (int i = 0; i < 8; ++i) acc[i] = vzero;

  const float* aRow   = adj + ((size_t)b * NNODES + r0 + n15) * NNODES;
  const float* gBbase = support + (size_t)b * NNODES * FDIM;

  for (int k0c = 0; k0c < NNODES; k0c += KB) {
    __syncthreads();                       // previous chunk fully consumed
    // ---- stage support chunk: fp32 -> bf16 hi/lo planes (convert once) ----
    const float* gB = gBbase + (size_t)k0c * FDIM;
#pragma unroll 2
    for (int idx = tid; idx < KB * FDIM / 4; idx += BLK_THREADS) {
      int kk = idx >> 5;                   // 0..KB-1
      int n0 = (idx & 31) * 4;             // column group (lane-contiguous)
      v4f v = *(const v4f*)(gB + kk * FDIM + n0);
#pragma unroll
      for (int j = 0; j < 4; ++j) {
        int n = n0 + j;
        int si = n * KB + (((kk >> 3) ^ (n & 15)) << 3) + (kk & 7);
        __bf16 h, l;
        split_bf16(v[j], h, l);
        sHI[si] = h;
        sLO[si] = l;
      }
    }
    __syncthreads();

    // ---- compute: KB/32 = 4 bf16 WMMA k-steps over this chunk ----
    for (int ks = 0; ks < KB / 32; ++ks) {
      // A frag: row m = lane&15; half 0 reads K{0..7,16..23}, half 1 {8..15,24..31}
      const float* ap = aRow + k0c + ks * 32 + 8 * half;
      v4f s0 = *(const v4f*)(ap + 0);
      v4f s1 = *(const v4f*)(ap + 4);
      v4f s2 = *(const v4f*)(ap + 16);
      v4f s3 = *(const v4f*)(ap + 20);
      V16U ahi, alo;
#pragma unroll
      for (int j = 0; j < 4; ++j) {
        split_bf16(s0[j], ahi.e[j],      alo.e[j]);
        split_bf16(s1[j], ahi.e[4 + j],  alo.e[4 + j]);
        split_bf16(s2[j], ahi.e[8 + j],  alo.e[8 + j]);
        split_bf16(s3[j], ahi.e[12 + j], alo.e[12 + j]);
      }

      const int sb0 = 4 * ks + half;       // first 8-K sub-block for this lane
#pragma unroll
      for (int nt = 0; nt < 8; ++nt) {
        const int n = nt * 16 + n15;
        const int base = n * KB;
        const int sw = n & 15;
        V16U bhi, blo;
        *(v8bf*)&bhi.e[0] = *(const v8bf*)&sHI[base + (((sb0 + 0) ^ sw) << 3)];
        *(v8bf*)&bhi.e[8] = *(const v8bf*)&sHI[base + (((sb0 + 2) ^ sw) << 3)];
        *(v8bf*)&blo.e[0] = *(const v8bf*)&sLO[base + (((sb0 + 0) ^ sw) << 3)];
        *(v8bf*)&blo.e[8] = *(const v8bf*)&sLO[base + (((sb0 + 2) ^ sw) << 3)];
        acc[nt] = __builtin_amdgcn_wmma_f32_16x16x32_bf16(
            false, ahi.v, false, bhi.v, (short)0, acc[nt], false, false);
        acc[nt] = __builtin_amdgcn_wmma_f32_16x16x32_bf16(
            false, alo.v, false, bhi.v, (short)0, acc[nt], false, false);
        acc[nt] = __builtin_amdgcn_wmma_f32_16x16x32_bf16(
            false, ahi.v, false, blo.v, (short)0, acc[nt], false, false);
      }
    }
  }

  // ---- bias + store (C layout: V[i] -> row i + 8*half, col = nt*16+n15) ----
  float* o = out + ((size_t)b * NNODES + r0) * FDIM;
#pragma unroll
  for (int nt = 0; nt < 8; ++nt) {
    const int col = nt * 16 + n15;
    const float bb = bias[col];
#pragma unroll
    for (int i = 0; i < 8; ++i)
      o[(i + 8 * half) * FDIM + col] = acc[nt][i] + bb;
  }
}

// ---------------------------------------------------------------------------
extern "C" void kernel_launch(void* const* d_in, const int* in_sizes, int n_in,
                              void* d_out, int out_size, void* d_ws, size_t ws_size,
                              hipStream_t stream) {
  (void)in_sizes; (void)n_in; (void)out_size; (void)ws_size;
  const float* x    = (const float*)d_in[0];   // [8, 2048, 128]
  const float* adj  = (const float*)d_in[1];   // [8, 2048, 2048]
  const float* W    = (const float*)d_in[2];   // [128, 128]
  const float* bias = (const float*)d_in[3];   // [128]
  float* out     = (float*)d_out;              // [8, 2048, 128]
  float* support = (float*)d_ws;               // 8 MB scratch

  const dim3 blk(BLK_THREADS);
  const dim3 grid(NB * (NNODES / ROWS_PB));    // 8 * 32 = 256 blocks

  hipLaunchKernelGGL(gcn_support_kernel, grid, blk, 0, stream, x, W, support);
  hipLaunchKernelGGL(gcn_aggregate_kernel, grid, blk, 0, stream,
                     adj, support, bias, out);
}
